// MoE_85478439125273
// MI455X (gfx1250) — matrix operand order repaired and checked by reading, not compiled, not run on hardware
//
#include <hip/hip_runtime.h>
#include <math.h>

// ---------------- config ----------------
#define DIM        1024
#define N_EXPERTS  16
#define MOE_H      512
#define SH_H       2048
#define TM         64      // tokens per workgroup tile (4 sub-tiles of 16)
#define THREADS    512     // 16 wave32s
#define NWAVE      8       // waves per router block

typedef _Float16 v4h  __attribute__((ext_vector_type(4)));
typedef _Float16 v8h  __attribute__((ext_vector_type(8)));
typedef _Float16 v16h __attribute__((ext_vector_type(16)));
typedef float    v4f  __attribute__((ext_vector_type(4)));
typedef float    v8f  __attribute__((ext_vector_type(8)));

static __device__ __forceinline__ v8f wmma16(v16h a, v16h b, v8f c) {
    // D(16x16,f32) = A(16x32,f16) * B(32x16,f16) + C(16x16,f32)
    return __builtin_amdgcn_wmma_f32_16x16x32_f16(
        /*neg_a*/false, a, /*neg_b*/false, b,
        /*c_mod*/(short)0, c, /*reuse_a*/false, /*reuse_b*/false);
}

// A fragment: 8 consecutive f16 at p, 8 consecutive at p+16 (ISA 16-bit A k-split)
static __device__ __forceinline__ v16h ldA(const _Float16* p) {
    v8h lo = *(const v8h*)p;
    v8h hh = *(const v8h*)(p + 16);
    return __builtin_shufflevector(lo, hh, 0,1,2,3,4,5,6,7,8,9,10,11,12,13,14,15);
}

// ================= Router kernel (exact f32) =================
__global__ void __launch_bounds__(256)
moe_router(const float* __restrict__ x, const float* __restrict__ gw,
           const float* __restrict__ be, float* __restrict__ combine) {
    const int tid  = threadIdx.x;
    const int wave = tid >> 5;
    const int lane = tid & 31;
    const int t    = blockIdx.x * NWAVE + wave;

    const float* xr = x + (size_t)t * DIM;
    float acc[N_EXPERTS];
#pragma unroll
    for (int e = 0; e < N_EXPERTS; ++e) acc[e] = 0.f;

    for (int d = lane; d < DIM; d += 32) {
        float xv = xr[d];
#pragma unroll
        for (int e = 0; e < N_EXPERTS; ++e)
            acc[e] = fmaf(xv, gw[e * DIM + d], acc[e]);
    }
#pragma unroll
    for (int e = 0; e < N_EXPERTS; ++e) {
#pragma unroll
        for (int off = 16; off > 0; off >>= 1)
            acc[e] += __shfl_xor(acc[e], off, 32);
    }

    float s0[N_EXPERTS], sc[N_EXPERTS];
#pragma unroll
    for (int e = 0; e < N_EXPERTS; ++e) {
        float s = 1.f / (1.f + expf(-acc[e]));
        s0[e] = s;              // weights use raw sigmoid scores
        sc[e] = s + be[e];      // selection uses bias-corrected scores
    }

    float gs[4];
#pragma unroll
    for (int g = 0; g < 4; ++g) {
        float t0 = sc[4 * g + 0], t1 = sc[4 * g + 1];
        float t2 = sc[4 * g + 2], t3 = sc[4 * g + 3];
        float a = fmaxf(t0, t1), b = fminf(t0, t1);
        float c = fmaxf(t2, t3), d2 = fminf(t2, t3);
        float m1 = fmaxf(a, c);
        float m2 = (a >= c) ? fmaxf(b, c) : fmaxf(d2, a);
        gs[g] = m1 + m2;
    }
    int g1 = 0;
#pragma unroll
    for (int g = 1; g < 4; ++g) if (gs[g] > gs[g1]) g1 = g;
    int g2 = (g1 == 0) ? 1 : 0;
#pragma unroll
    for (int g = 0; g < 4; ++g) if (g != g1 && gs[g] > gs[g2]) g2 = g;

    float msk[N_EXPERTS];
#pragma unroll
    for (int e = 0; e < N_EXPERTS; ++e) {
        int g = e >> 2;
        msk[e] = (g == g1 || g == g2) ? sc[e] : 0.0f;
    }
    int   idx[4];
    float tw[4];
#pragma unroll
    for (int k = 0; k < 4; ++k) {
        int bi = 0;
#pragma unroll
        for (int e = 1; e < N_EXPERTS; ++e) if (msk[e] > msk[bi]) bi = e;
        idx[k] = bi;
        float sv = 0.f;
#pragma unroll
        for (int e = 0; e < N_EXPERTS; ++e) sv = (e == bi) ? s0[e] : sv;
        tw[k] = sv;
#pragma unroll
        for (int e = 0; e < N_EXPERTS; ++e) msk[e] = (e == bi) ? -1e30f : msk[e];
    }
    float rs = 2.5f / (tw[0] + tw[1] + tw[2] + tw[3] + 1e-20f);

    if (lane < N_EXPERTS) {
        float w = 0.f;
#pragma unroll
        for (int k = 0; k < 4; ++k) w += (idx[k] == lane) ? tw[k] * rs : 0.f;
        combine[(size_t)t * N_EXPERTS + lane] = w;
    }
}

// ================= Fused MoE kernel (f16 WMMA, f32 accumulate) =================
__global__ void __launch_bounds__(THREADS)
moe_fused(const float* __restrict__ x,
          const float* __restrict__ up_w, const float* __restrict__ down_w,
          const float* __restrict__ shared_up, const float* __restrict__ shared_down,
          const float* __restrict__ combine, float* __restrict__ out) {
    extern __shared__ char smem[];
    _Float16* xs    = (_Float16*)smem;                       // 64*1024 f16 = 128 KB
    _Float16* wslab = xs + TM * DIM;                         // 32768 f16  =  64 KB
    _Float16* hbuf  = wslab + 32768;                         // 64*512 f16 =  64 KB
    float*    cws   = (float*)(hbuf + TM * MOE_H);           // 64*16 f32  =   4 KB

    const int tid    = threadIdx.x;
    const int wave   = tid >> 5;
    const int lane   = tid & 31;
    const int lane15 = lane & 15;
    const int hi     = lane >> 4;        // half-wave select (contiguous k-split: k = 16*hi + s)
    const int msub   = wave & 3;         // M sub-tile (16 tokens each)
    const int nquad  = wave >> 2;        // N quadrant

    const int token0 = blockIdx.x * TM;

    // ---- load + convert x tile: 4 batches of 8 in-flight b128 loads ----
    {
        const float* src = x + (size_t)token0 * DIM;
#pragma unroll
        for (int half = 0; half < 4; ++half) {
            v4f t[8];
#pragma unroll
            for (int i = 0; i < 8; ++i) {
                int fl = (half * 8 + i) * (THREADS * 4) + tid * 4;
                t[i] = *(const v4f*)&src[fl];
            }
#pragma unroll
            for (int i = 0; i < 8; ++i) {
                int fl = (half * 8 + i) * (THREADS * 4) + tid * 4;
                v4h h = { (_Float16)t[i].x, (_Float16)t[i].y, (_Float16)t[i].z, (_Float16)t[i].w };
                *(v4h*)&xs[fl] = h;
            }
        }
#pragma unroll
        for (int i = 0; i < 2; ++i)
            cws[i * THREADS + tid] = combine[(size_t)token0 * N_EXPERTS + i * THREADS + tid];
    }
    __syncthreads();

    v8f facc[16];
#pragma unroll
    for (int j = 0; j < 16; ++j) facc[j] = (v8f){0.f,0.f,0.f,0.f,0.f,0.f,0.f,0.f};

    const int arow = (msub * 16 + lane15) * DIM;       // A-row base in xs
    const int hrow = (msub * 16 + lane15) * MOE_H;     // A-row base in hbuf (down)

    for (int unit = 0; unit < N_EXPERTS + 4; ++unit) {
        const float* ub;
        const float* db;
        int dstride;
        bool is_exp = (unit < N_EXPERTS);
        if (is_exp) {
            ub = up_w   + (size_t)unit * MOE_H * DIM;
            db = down_w + (size_t)unit * DIM * MOE_H;
            dstride = MOE_H;
        } else {
            int c = unit - N_EXPERTS;                     // shared chunk of 512
            ub = shared_up   + (size_t)c * MOE_H * DIM;   // rows [512c, 512c+512)
            db = shared_down + (size_t)c * MOE_H;         // k-columns [512c, ...)
            dstride = SH_H;
        }

        // ---------- UP: h = cw * relu2(x_tile @ Wu^T), M=64 N=512 K=1024 ----------
        v8f hacc[8];
#pragma unroll
        for (int j = 0; j < 8; ++j) hacc[j] = (v8f){0.f,0.f,0.f,0.f,0.f,0.f,0.f,0.f};

        for (int kb = 0; kb < 16; ++kb) {        // K-block of 64
            const int k0 = kb * 64;
            __syncthreads();                     // protect slab from prior readers
#pragma unroll
            for (int half = 0; half < 2; ++half) {   // stage Wu[0:512][k0:k0+64] as f16
                v4f t[8];
#pragma unroll
                for (int i = 0; i < 8; ++i) {
                    int fl = (half * 8 + i) * (THREADS * 4) + tid * 4;
                    int n = fl >> 6, kk = fl & 63;
                    t[i] = *(const v4f*)&ub[(size_t)n * DIM + k0 + kk];
                }
#pragma unroll
                for (int i = 0; i < 8; ++i) {
                    int fl = (half * 8 + i) * (THREADS * 4) + tid * 4;
                    int n = fl >> 6, kk = fl & 63;
                    v4h h = { (_Float16)t[i].x, (_Float16)t[i].y, (_Float16)t[i].z, (_Float16)t[i].w };
                    *(v4h*)&wslab[n * 64 + kk] = h;
                }
            }
            __syncthreads();
#pragma unroll
            for (int ks = 0; ks < 2; ++ks) {     // two K=32 WMMA steps per block
                const int kk0 = ks * 32;
                v16h a = *(const v16h*)&xs[arow + k0 + kk0 + hi * 16];
                const _Float16* bbase = &wslab[(nquad * 128 + lane15) * 64 + kk0 + hi * 16];
                v16h bcur = *(const v16h*)bbase;           // j=0 fragment
#pragma unroll
                for (int j = 0; j < 8; ++j) {              // double-buffered B
                    v16h bnext = bcur;
                    if (j < 7) bnext = *(const v16h*)(bbase + (j + 1) * (16 * 64));
                    hacc[j] = wmma16(a, bcur, hacc[j]);
                    bcur = bnext;
                }
            }
        }
        // relu^2, fold combine weight, store f16 -> hbuf (C layout: M=r+8*hi, N=lane15)
        __syncthreads();
#pragma unroll
        for (int r = 0; r < 8; ++r) {
            int m = msub * 16 + r + hi * 8;
            float cw = is_exp ? cws[m * N_EXPERTS + unit] : 1.0f;
#pragma unroll
            for (int j = 0; j < 8; ++j) {
                int n = nquad * 128 + j * 16 + lane15;
                float v = fmaxf(hacc[j][r], 0.f);
                hbuf[m * MOE_H + n] = (_Float16)(cw * v * v);
            }
        }
        __syncthreads();

        // ---------- DOWN: facc += h @ Wd^T, M=64 N=1024 K=512 ----------
        for (int kb = 0; kb < 16; ++kb) {        // K-block of 32
            const int k0 = kb * 32;
            __syncthreads();
#pragma unroll
            for (int half = 0; half < 2; ++half) {   // stage Wd[0:1024][k0:k0+32] as f16
                v4f t[8];
#pragma unroll
                for (int i = 0; i < 8; ++i) {
                    int fl = (half * 8 + i) * (THREADS * 4) + tid * 4;
                    int n = fl >> 5, kk = fl & 31;
                    t[i] = *(const v4f*)&db[(size_t)n * dstride + k0 + kk];
                }
#pragma unroll
                for (int i = 0; i < 8; ++i) {
                    int fl = (half * 8 + i) * (THREADS * 4) + tid * 4;
                    int n = fl >> 5, kk = fl & 31;
                    v4h h = { (_Float16)t[i].x, (_Float16)t[i].y, (_Float16)t[i].z, (_Float16)t[i].w };
                    *(v4h*)&wslab[n * 32 + kk] = h;
                }
            }
            __syncthreads();
            v16h a = *(const v16h*)&hbuf[hrow + k0 + hi * 16];
            const _Float16* bbase = &wslab[(nquad * 256 + lane15) * 32 + hi * 16];
            v16h bcur = *(const v16h*)bbase;               // j=0 fragment
#pragma unroll
            for (int j = 0; j < 16; ++j) {                 // double-buffered B
                v16h bnext = bcur;
                if (j < 15) bnext = *(const v16h*)(bbase + (j + 1) * (16 * 32));
                facc[j] = wmma16(a, bcur, facc[j]);        // cw already folded into h
                bcur = bnext;
            }
        }
    }

    // ---- store output tile ----
#pragma unroll
    for (int j = 0; j < 16; ++j) {
        int n = nquad * 256 + j * 16 + lane15;
#pragma unroll
        for (int r = 0; r < 8; ++r) {
            int m = msub * 16 + r + hi * 8;
            out[(size_t)(token0 + m) * DIM + n] = facc[j][r];
        }
    }
}

// ================= launch =================
extern "C" void kernel_launch(void* const* d_in, const int* in_sizes, int n_in,
                              void* d_out, int out_size, void* d_ws, size_t ws_size,
                              hipStream_t stream) {
    const float* x    = (const float*)d_in[0];  // [2,2048,1024]
    const float* gw   = (const float*)d_in[1];  // [16,1024]
    const float* be   = (const float*)d_in[2];  // [16]
    const float* up   = (const float*)d_in[3];  // [16,512,1024]
    const float* dn   = (const float*)d_in[4];  // [16,1024,512]
    const float* sup  = (const float*)d_in[5];  // [2048,1024]
    const float* sdn  = (const float*)d_in[6];  // [1024,2048]
    float*       outp = (float*)d_out;          // [2,2048,1024]
    float*       combine = (float*)d_ws;        // [4096,16] f32 = 256 KB scratch

    const int T = 4096;
    moe_router<<<T / NWAVE, 256, 0, stream>>>(x, gw, be, combine);

    // LDS: x tile 128KB + weight slab 64KB + h buffer 64KB + combine 4KB = 260KB
    const size_t smem_bytes =
        (size_t)(TM * DIM + 32768 + TM * MOE_H) * sizeof(_Float16)
        + (size_t)(TM * N_EXPERTS) * sizeof(float);  // 266240 bytes
    moe_fused<<<T / TM, THREADS, smem_bytes, stream>>>(x, up, dn, sup, sdn, combine, outp);
}
